// MLPBlock_64115271794781
// MI455X (gfx1250) — matrix operand order repaired and checked by reading, not compiled, not run on hardware
//
#include <hip/hip_runtime.h>
#include <hip/hip_bf16.h>
#include <cstdint>
#include <cstddef>

// Problem sizes (fixed by the reference)
#define H_DIM 1024
#define I_DIM 4096
#define E_NUM 16
#define T_NUM 4096
#define NSLOT (2 * T_NUM)   // top_k = 2 -> slot id = token*2 + k

typedef __attribute__((ext_vector_type(16))) __bf16 v16bf;
typedef __attribute__((ext_vector_type(8)))  float  v8f;

union FragBF {
  v16bf v;
  unsigned short u[16];
  uint4 q4[2];
};

// Native fptrunc (RNE on AMDGPU) -> hardware cvt path instead of bit-twiddles.
__device__ __forceinline__ unsigned short f32_to_bf16(float f) {
  return __builtin_bit_cast(unsigned short, (__bf16)f);
}
__device__ __forceinline__ unsigned pack_bf16x2(float lo, float hi) {
  return (unsigned)f32_to_bf16(lo) | ((unsigned)f32_to_bf16(hi) << 16);
}

// CDNA5 async global->LDS copy (ASYNCcnt-tracked, bypasses VGPRs).
// Per-lane: LDS[ldsAddr .. +15] = MEM[gaddr .. +15].
__device__ __forceinline__ void async_copy_b128(unsigned ldsAddr,
                                                const void* gaddr) {
  asm volatile("global_load_async_to_lds_b128 %0, %1, off"
               :
               : "v"(ldsAddr), "v"(gaddr)
               : "memory");
}
__device__ __forceinline__ void wait_asynccnt0() {
  asm volatile("s_wait_asynccnt 0" ::: "memory");
}

// ---------------------------------------------------------------------------
// Kernel 1: router.  One wave32 per token: 16 dot products over H=1024,
// cross-lane reduction, top-2 + softmax.  Writes per-slot expert id + weight.
// ---------------------------------------------------------------------------
__global__ __launch_bounds__(256) void moe_router(
    const float* __restrict__ x, const float* __restrict__ Wr,
    const float* __restrict__ br, int* __restrict__ topk_idx,
    float* __restrict__ wt_slot) {
  const int lane = threadIdx.x & 31;
  const int wv   = threadIdx.x >> 5;
  const int t    = blockIdx.x * 8 + wv;
  float acc[E_NUM];
#pragma unroll
  for (int e = 0; e < E_NUM; ++e) acc[e] = 0.f;
  for (int h = lane; h < H_DIM; h += 32) {
    const float xv = x[(size_t)t * H_DIM + h];
#pragma unroll
    for (int e = 0; e < E_NUM; ++e) acc[e] += xv * Wr[e * H_DIM + h];
  }
#pragma unroll
  for (int e = 0; e < E_NUM; ++e) {
#pragma unroll
    for (int off = 16; off > 0; off >>= 1)
      acc[e] += __shfl_xor(acc[e], off, 32);
  }
  if (lane == 0) {
    float b0 = -1e30f, b1 = -1e30f;
    int   i0 = 0,      i1 = 0;
#pragma unroll
    for (int e = 0; e < E_NUM; ++e) {
      const float v = acc[e] + br[e];
      if (v > b0)      { b1 = b0; i1 = i0; b0 = v; i0 = e; }
      else if (v > b1) { b1 = v;  i1 = e; }
    }
    const float e1  = __expf(b1 - b0);         // softmax over the two picks
    const float inv = 1.f / (1.f + e1);
    topk_idx[t * 2 + 0] = i0;
    topk_idx[t * 2 + 1] = i1;
    wt_slot[t * 2 + 0]  = inv;
    wt_slot[t * 2 + 1]  = e1 * inv;
  }
}

// ---------------------------------------------------------------------------
// Kernel 2: build per-expert slot lists deterministically (prefix scan, no
// atomics -> stable ordering -> bit-stable float accumulation downstream).
// ---------------------------------------------------------------------------
__global__ __launch_bounds__(256) void moe_build_lists(
    const int* __restrict__ topk_idx, int* __restrict__ counts,
    int* __restrict__ slotlist) {
  __shared__ int scan[256];
  __shared__ int base;
  const int e = blockIdx.x, tid = threadIdx.x;
  if (tid == 0) base = 0;
  __syncthreads();
  for (int s0 = 0; s0 < NSLOT; s0 += 256) {
    const int s    = s0 + tid;
    const int flag = (topk_idx[s] == e) ? 1 : 0;
    scan[tid] = flag;
    __syncthreads();
    for (int off = 1; off < 256; off <<= 1) {   // Hillis-Steele inclusive scan
      const int v = (tid >= off) ? scan[tid - off] : 0;
      __syncthreads();
      scan[tid] += v;
      __syncthreads();
    }
    if (flag) slotlist[e * T_NUM + base + scan[tid] - 1] = s;
    __syncthreads();
    if (tid == 0) base += scan[255];
    __syncthreads();
  }
  if (tid == 0) counts[e] = base;
}

// ---------------------------------------------------------------------------
// Kernel 3: x fp32 -> bf16.
// ---------------------------------------------------------------------------
__global__ __launch_bounds__(256) void moe_cvt_x(
    const float* __restrict__ x, unsigned short* __restrict__ xbf) {
  const int idx = (blockIdx.x * 256 + threadIdx.x) * 4;
  const float4 v = *(const float4*)(x + idx);
  uint2 p;
  p.x = pack_bf16x2(v.x, v.y);
  p.y = pack_bf16x2(v.z, v.w);
  *(uint2*)(xbf + idx) = p;
}

// ---------------------------------------------------------------------------
// Kernel 4: grouped GEMM1 + SwiGLU.  Block = 64 gathered token-slots x
// (32 gate + 32 up) cols of expert blockIdx.z, K over H in steps of 32.
// Double-buffered LDS pipeline:
//   iter i: wait async(tile i) -> barrier -> issue tile i+1 (async A DMA +
//           coalesced fp32 weight loads converted to bf16, stored transposed)
//           -> ds_load_b128 fragments of tile i -> 2x WMMA.
// ---------------------------------------------------------------------------
__global__ __launch_bounds__(256) void moe_gemm1(
    const unsigned short* __restrict__ xbf, const float* __restrict__ w13,
    const int* __restrict__ counts, const int* __restrict__ slotlist,
    unsigned short* __restrict__ act) {
  const int e   = blockIdx.z;
  const int cnt = counts[e];
  const int rowbase = blockIdx.x * 64;
  if (rowbase >= cnt) return;                 // uniform early-exit
  const int gcol = blockIdx.y * 32;           // gate column base within I

  __shared__ unsigned short As[2][64][40];    // row-major tokens x K
  __shared__ unsigned short Bg[2][32][40];    // transposed: [n][k]
  __shared__ unsigned short Bu[2][32][40];    // transposed: [n][k]
  __shared__ int s_slot[64];

  const int tid = threadIdx.x;
  if (tid < 64) {
    const int g = rowbase + tid;
    s_slot[tid] = (g < cnt) ? slotlist[e * T_NUM + g] : -1;
  }
  __syncthreads();

  // A-tile copy mapping: 256 threads = 64 rows x 4 chunks of 8 bf16 (16B)
  const int ar = tid >> 2;
  const int acnk = (tid & 3) * 8;
  const int aslot = s_slot[ar];
  const unsigned ldsA[2] = {(unsigned)(uintptr_t)&As[0][ar][acnk],
                            (unsigned)(uintptr_t)&As[1][ar][acnk]};
  const unsigned short* agsrc =
      (aslot >= 0) ? (xbf + (size_t)(aslot >> 1) * H_DIM + acnk) : nullptr;
  if (aslot < 0) {                            // zero pad rows once; DMA skips them
    uint4 z = {};
    *(uint4*)&As[0][ar][acnk] = z;
    *(uint4*)&As[1][ar][acnk] = z;
  }
  // B-tile mapping: lane = column n (coalesced 128B rows), wave = 4 K-rows
  const int bn = tid & 31;
  const int bk = (tid >> 5) * 4;
  const float* w13e = w13 + (size_t)e * H_DIM * (2 * I_DIM);

  auto issue_tile = [&](int kk, int buf) {
    if (aslot >= 0) async_copy_b128(ldsA[buf], agsrc + kk);
    const float* pg = w13e + (size_t)(kk + bk) * (2 * I_DIM) + gcol + bn;
    const float* pu = pg + I_DIM;
    float g0 = pg[0];
    float g1 = pg[2 * I_DIM];
    float g2 = pg[4 * I_DIM];
    float g3 = pg[6 * I_DIM];
    float u0 = pu[0];
    float u1 = pu[2 * I_DIM];
    float u2 = pu[4 * I_DIM];
    float u3 = pu[6 * I_DIM];
    uint2 gv, uv;
    gv.x = pack_bf16x2(g0, g1);
    gv.y = pack_bf16x2(g2, g3);
    uv.x = pack_bf16x2(u0, u1);
    uv.y = pack_bf16x2(u2, u3);
    *(uint2*)&Bg[buf][bn][bk] = gv;
    *(uint2*)&Bu[buf][bn][bk] = uv;
  };

  const int lane = tid & 31;
  const int w    = tid >> 5;
  const int rsub = w & 3;                     // 4 row sub-tiles of 16
  const int csub = w >> 2;                    // 2 col sub-tiles of 16
  const int m    = rsub * 16 + (lane & 15);
  const int kbA  = (lane < 16) ? 0 : 8;
  const int n_l  = csub * 16 + (lane & 15);
  const int kbB  = (lane < 16) ? 0 : 8;
  v8f accg = {}; v8f accu = {};

  issue_tile(0, 0);                           // prologue: fill buffer 0

#pragma unroll 2
  for (int kk = 0; kk < H_DIM; kk += 32) {
    const int ib = (kk >> 5) & 1;
    wait_asynccnt0();                         // tile i's DMA complete (per wave)
    __syncthreads();                          // all waves' tile-i writes visible
    if (kk + 32 < H_DIM) issue_tile(kk + 32, ib ^ 1);  // overlap with WMMA

    FragBF a, bg, bu;
    a.q4[0]  = *(const uint4*)&As[ib][m][kbA];          // K kb..kb+7
    a.q4[1]  = *(const uint4*)&As[ib][m][kbA + 16];     // K kb+16..kb+23
    bg.q4[0] = *(const uint4*)&Bg[ib][n_l][kbB];        // K kb..kb+7
    bg.q4[1] = *(const uint4*)&Bg[ib][n_l][kbB + 16];   // K kb+16..kb+23
    bu.q4[0] = *(const uint4*)&Bu[ib][n_l][kbB];
    bu.q4[1] = *(const uint4*)&Bu[ib][n_l][kbB + 16];
    accg = __builtin_amdgcn_wmma_f32_16x16x32_bf16(false, a.v, false, bg.v,
                                                   (short)0, accg, false, false);
    accu = __builtin_amdgcn_wmma_f32_16x16x32_bf16(false, a.v, false, bu.v,
                                                   (short)0, accu, false, false);
  }

  // Fused SwiGLU epilogue -> bf16 act[slot, col]
  const int col = gcol + csub * 16 + (lane & 15);
  const int mb  = rsub * 16 + ((lane < 16) ? 0 : 8);
#pragma unroll
  for (int i = 0; i < 8; ++i) {
    const int r    = mb + i;
    const int slot = s_slot[r];
    if (slot >= 0) {
      const float g = accg[i], u = accu[i];
      const float a = (g / (1.f + __expf(-g))) * u;   // silu(g) * u
      act[(size_t)slot * I_DIM + col] = f32_to_bf16(a);
    }
  }
}

// ---------------------------------------------------------------------------
// Kernel 5: grouped GEMM2 + weighted scatter.  out[t] += wt * act[slot] @ w2[e]
// 64 slots x 32 H-cols per block, K over I=4096.  Same double-buffered
// pipeline as GEMM1.  Exactly two atomic contributions per output element
// (commutative add -> deterministic).
// ---------------------------------------------------------------------------
__global__ __launch_bounds__(256) void moe_gemm2(
    const unsigned short* __restrict__ act, const float* __restrict__ w2,
    const int* __restrict__ counts, const int* __restrict__ slotlist,
    const float* __restrict__ wt_slot, float* __restrict__ out) {
  const int e   = blockIdx.z;
  const int cnt = counts[e];
  const int rowbase = blockIdx.x * 64;
  if (rowbase >= cnt) return;
  const int ncol = blockIdx.y * 32;           // H column base

  __shared__ unsigned short As[2][64][40];
  __shared__ unsigned short Bs[2][32][40];    // transposed [n][k]
  __shared__ int   s_slot[64];
  __shared__ int   s_tok[64];
  __shared__ float s_wt[64];

  const int tid = threadIdx.x;
  if (tid < 64) {
    const int g    = rowbase + tid;
    const int slot = (g < cnt) ? slotlist[e * T_NUM + g] : -1;
    s_slot[tid] = slot;
    s_tok[tid]  = (slot >= 0) ? (slot >> 1) : 0;
    s_wt[tid]   = (slot >= 0) ? wt_slot[slot] : 0.f;
  }
  __syncthreads();

  const int ar = tid >> 2;
  const int acnk = (tid & 3) * 8;
  const int aslot = s_slot[ar];
  const unsigned ldsA[2] = {(unsigned)(uintptr_t)&As[0][ar][acnk],
                            (unsigned)(uintptr_t)&As[1][ar][acnk]};
  const unsigned short* agsrc =
      (aslot >= 0) ? (act + (size_t)aslot * I_DIM + acnk) : nullptr;
  if (aslot < 0) {
    uint4 z = {};
    *(uint4*)&As[0][ar][acnk] = z;
    *(uint4*)&As[1][ar][acnk] = z;
  }
  const int bn = tid & 31;
  const int bk = (tid >> 5) * 4;
  const float* w2e = w2 + (size_t)e * I_DIM * H_DIM;

  auto issue_tile = [&](int kk, int buf) {
    if (aslot >= 0) async_copy_b128(ldsA[buf], agsrc + kk);
    const float* pb = w2e + (size_t)(kk + bk) * H_DIM + ncol + bn;
    float b0 = pb[0];
    float b1 = pb[H_DIM];
    float b2 = pb[2 * H_DIM];
    float b3 = pb[3 * H_DIM];
    uint2 bv;
    bv.x = pack_bf16x2(b0, b1);
    bv.y = pack_bf16x2(b2, b3);
    *(uint2*)&Bs[buf][bn][bk] = bv;
  };

  const int lane = tid & 31;
  const int w    = tid >> 5;
  const int rsub = w & 3;
  const int csub = w >> 2;
  const int m    = rsub * 16 + (lane & 15);
  const int kbA  = (lane < 16) ? 0 : 8;
  const int n_l  = csub * 16 + (lane & 15);
  const int kbB  = (lane < 16) ? 0 : 8;
  v8f acc = {};

  issue_tile(0, 0);

#pragma unroll 2
  for (int kk = 0; kk < I_DIM; kk += 32) {
    const int ib = (kk >> 5) & 1;
    wait_asynccnt0();
    __syncthreads();
    if (kk + 32 < I_DIM) issue_tile(kk + 32, ib ^ 1);

    FragBF a, b;
    a.q4[0] = *(const uint4*)&As[ib][m][kbA];
    a.q4[1] = *(const uint4*)&As[ib][m][kbA + 16];
    b.q4[0] = *(const uint4*)&Bs[ib][n_l][kbB];
    b.q4[1] = *(const uint4*)&Bs[ib][n_l][kbB + 16];
    acc = __builtin_amdgcn_wmma_f32_16x16x32_bf16(false, a.v, false, b.v,
                                                  (short)0, acc, false, false);
  }

  const int col = ncol + csub * 16 + (lane & 15);
  const int mb  = rsub * 16 + ((lane < 16) ? 0 : 8);
#pragma unroll
  for (int i = 0; i < 8; ++i) {
    const int r = mb + i;
    if (s_slot[r] >= 0)
      atomicAdd(&out[(size_t)s_tok[r] * H_DIM + col], s_wt[r] * acc[i]);
  }
}

// ---------------------------------------------------------------------------
// Launcher.  Workspace layout (bytes):
//   [0,32K)      topk_idx  int[8192]
//   [32K,64K)    wt_slot   float[8192]
//   [64K,64K+64) counts    int[16]
//   [128K,384K)  slotlist  int[16][4096]
//   [1M,9M)      x bf16    [4096][1024]
//   [16M,80M)    act bf16  [8192][4096]
// ---------------------------------------------------------------------------
extern "C" void kernel_launch(void* const* d_in, const int* in_sizes, int n_in,
                              void* d_out, int out_size, void* d_ws,
                              size_t ws_size, hipStream_t stream) {
  const float* x   = (const float*)d_in[0];
  const float* Wr  = (const float*)d_in[1];
  const float* br  = (const float*)d_in[2];
  const float* w13 = (const float*)d_in[3];
  const float* w2  = (const float*)d_in[4];
  // d_in[5] = top_k scalar; fixed to 2 in this kernel.
  float* out = (float*)d_out;

  char* ws = (char*)d_ws;
  int*            topk_idx = (int*)(ws + 0);
  float*          wt_slot  = (float*)(ws + (32 << 10));
  int*            counts   = (int*)(ws + (64 << 10));
  int*            slotlist = (int*)(ws + (128 << 10));
  unsigned short* xbf      = (unsigned short*)(ws + (1 << 20));
  unsigned short* act      = (unsigned short*)(ws + (size_t)(16 << 20));

  hipMemsetAsync(out, 0, sizeof(float) * (size_t)T_NUM * H_DIM, stream);

  moe_router<<<T_NUM / 8, 256, 0, stream>>>(x, Wr, br, topk_idx, wt_slot);
  moe_build_lists<<<E_NUM, 256, 0, stream>>>(topk_idx, counts, slotlist);
  moe_cvt_x<<<(T_NUM * H_DIM) / 1024, 256, 0, stream>>>(x, xbf);

  dim3 g1(T_NUM / 64, I_DIM / 32, E_NUM);
  moe_gemm1<<<g1, 256, 0, stream>>>(xbf, w13, counts, slotlist, act);

  dim3 g2(T_NUM / 64, H_DIM / 32, E_NUM);
  moe_gemm2<<<g2, 256, 0, stream>>>(act, w2, counts, slotlist, wt_slot, out);
}
